// SelfAttention_86114094284854
// MI455X (gfx1250) — matrix operand order repaired and checked by reading, not compiled, not run on hardware
//
#include <hip/hip_runtime.h>
#include <hip/hip_bf16.h>

typedef __attribute__((ext_vector_type(16))) _Float16 v16h;
typedef __attribute__((ext_vector_type(8)))  float    v8f;

union F16x16 { v16h v; uint4 q[2]; _Float16 h[16]; unsigned u[8]; };
union F16x8  { uint4 q; _Float16 h[8]; unsigned u[4]; };
union F32x8  { v8f v; float f[8]; };

constexpr int NB = 4, LSEQ = 2048, EMB = 1024, NH = 16, HD = 64;

// A-operand fragment: lane l holds row (l&15); 16 contiguous K values (low lanes K 0..15, high lanes 16..31)
__device__ __forceinline__ v16h frag_a16(const _Float16* p) {
  F16x16 f;
  f.q[0] = *(const uint4*)(p);
  f.q[1] = *(const uint4*)(p + 8);
  return f.v;
}
// B-operand fragment: lane n holds one column, 16 contiguous K values (low lanes K 0..15, high lanes 16..31)
__device__ __forceinline__ v16h frag_b16(const _Float16* p) {
  F16x16 f;
  f.q[0] = *(const uint4*)(p);
  f.q[1] = *(const uint4*)(p + 8);
  return f.v;
}
__device__ __forceinline__ v8f wmma_f16(v16h a, v16h b, v8f c) {
  return __builtin_amdgcn_wmma_f32_16x16x32_f16(false, a, false, b, (short)0, c, false, false);
}

// CDNA5 async global->LDS copy (ASYNCcnt-tracked, no VGPR data path).
// VGLOBAL encoding: VDST = LDS byte offset VGPR, VADDR = 64-bit global address, SADDR=off.
// Generic shared pointers carry the LDS offset in their low 32 bits (aperture rule).
__device__ __forceinline__ void async_ld_b128(void* lds_ptr, const void* gptr) {
  unsigned lds_off = (unsigned)(uintptr_t)lds_ptr;
  unsigned long long ga = (unsigned long long)(uintptr_t)gptr;
  asm volatile("global_load_async_to_lds_b128 %0, %1, off"
               :: "v"(lds_off), "v"(ga) : "memory");
}
__device__ __forceinline__ void wait_async0() {
#if __has_builtin(__builtin_amdgcn_s_wait_asynccnt)
  __builtin_amdgcn_s_wait_asynccnt(0);
#else
  asm volatile("s_wait_asynccnt 0x0" ::: "memory");
#endif
}

// ---------------------------------------------------------------- weights fp32 -> f16
__global__ __launch_bounds__(256) void cvt_weights(const float* Wv, const float* Wk,
                                                   const float* Wq, const float* Wo,
                                                   _Float16* Wv16, _Float16* Wk16,
                                                   _Float16* Wq16, _Float16* Wo16) {
  int i = blockIdx.x * 256 + threadIdx.x;
  if (i < EMB * EMB) Wo16[i] = (_Float16)Wo[i];
  if (i < HD * HD) {
    Wv16[i] = (_Float16)Wv[i];
    Wk16[i] = (_Float16)Wk[i];
    Wq16[i] = (_Float16)Wq[i];
  }
}

// ---------------------------------------------------------------- per-head QKV projection
// D(e,l) = W(16e x 32d) * X^T(32d x 16l); Qp/Kp stored [N,H,L,D], V stored transposed [N,H,D,L]
__global__ __launch_bounds__(256) void proj_qkv(const float* Xq, const float* Xk, const float* Xv,
                                                const _Float16* Wq16, const _Float16* Wk16,
                                                const _Float16* Wv16,
                                                _Float16* Qp, _Float16* Kp, _Float16* Vt) {
  const int lane = threadIdx.x & 31;
  const int wave = threadIdx.x >> 5;
  const int task = blockIdx.x * 8 + wave;     // 8192 = 512 l-tiles * 16 heads
  const int ltile = task >> 4;
  const int head  = task & 15;
  const int hi8  = (lane >> 4) << 3;
  const int hi16 = (lane >> 4) << 4;

  const int gl = ltile * 16 + (lane & 15);    // flat row in N*L
  const int nb = gl >> 11;                    // / 2048
  const int lr = gl & 2047;

  #pragma unroll
  for (int mat = 0; mat < 3; ++mat) {
    const float*    src = (mat == 0) ? Xq : (mat == 1) ? Xk : Xv;
    const _Float16* W   = (mat == 0) ? Wq16 : (mat == 1) ? Wk16 : Wv16;

    // B fragments: lane n holds x row gl, 16 contiguous d values (converted f32->f16)
    const float* xp = src + (size_t)gl * EMB + head * HD;
    F16x16 bf[2];
    #pragma unroll
    for (int dc = 0; dc < 2; ++dc) {
      const float* p = xp + dc * 32 + hi16;
      #pragma unroll
      for (int j = 0; j < 16; j += 4) {
        float4 f = *(const float4*)(p + j);
        bf[dc].h[j + 0] = (_Float16)f.x;
        bf[dc].h[j + 1] = (_Float16)f.y;
        bf[dc].h[j + 2] = (_Float16)f.z;
        bf[dc].h[j + 3] = (_Float16)f.w;
      }
    }

    v8f acc[4] = {};
    #pragma unroll
    for (int et = 0; et < 4; ++et) {
      #pragma unroll
      for (int dc = 0; dc < 2; ++dc) {
        v16h wa = frag_a16(W + (size_t)(et * 16 + (lane & 15)) * HD + dc * 32 + hi16);
        acc[et] = wmma_f16(wa, bf[dc].v, acc[et]);
      }
    }

    if (mat == 2) {
      // V transposed: rows = d (register dim), cols = l (lane) -> strided b16 stores
      #pragma unroll
      for (int et = 0; et < 4; ++et) {
        F32x8 u; u.v = acc[et];
        const int d0 = et * 16 + hi8;
        #pragma unroll
        for (int r = 0; r < 8; ++r)
          Vt[((size_t)((nb * NH + head) * HD) + d0 + r) * LSEQ + lr] = (_Float16)u.f[r];
      }
    } else {
      _Float16* dst = (mat == 0) ? Qp : Kp;
      const float sc = (mat == 0) ? 0.03125f : 1.0f;   // fold 1/sqrt(EMBED) into Q
      #pragma unroll
      for (int et = 0; et < 4; ++et) {
        F32x8 u; u.v = acc[et];
        F16x8 s;
        #pragma unroll
        for (int r = 0; r < 8; ++r) s.h[r] = (_Float16)(u.f[r] * sc);
        *(uint4*)&dst[((size_t)(nb * NH + head) * LSEQ + lr) * HD + et * 16 + hi8] = s.q;
      }
    }
  }
}

// ---------------------------------------------------------------- flash attention (S^T trick)
// K/V tiles stream into double-buffered LDS via global_load_async_to_lds_b128 (ASYNCcnt),
// overlapping the next tile's HBM traffic with the current tile's WMMAs + softmax.
__global__ __launch_bounds__(256) void flash_attn(const _Float16* Qp, const _Float16* Kp,
                                                  const _Float16* Vt, _Float16* O) {
  __shared__ _Float16 ldsK[2][32 * 64];   // 32 kv rows x 64 d, double buffered
  __shared__ _Float16 ldsV[2][64 * 32];   // 64 d rows x 32 kv (V already transposed)
  const int lane = threadIdx.x & 31;
  const int wave = threadIdx.x >> 5;
  const int head = blockIdx.y, nb = blockIdx.z;
  const int qbase = blockIdx.x * 128 + wave * 16;
  const int hi8  = (lane >> 4) << 3;
  const int hi16 = (lane >> 4) << 4;

  const _Float16* Qh = Qp + (size_t)(nb * NH + head) * LSEQ * HD;
  const _Float16* Kh = Kp + (size_t)(nb * NH + head) * LSEQ * HD;
  const _Float16* Vh = Vt + (size_t)(nb * NH + head) * HD * LSEQ;

  // cooperative staging: each of the 256 threads owns one 16B slot of K and of V
  const int t  = threadIdx.x;
  const int kr = t >> 3, kc = (t & 7) << 3;   // K tile: 32 x 64
  const int vr = t >> 2, vc = (t & 3) << 3;   // V tile: 64 x 32

  // prologue: kick off chunk 0 into buffer 0
  async_ld_b128(&ldsK[0][kr * 64 + kc], &Kh[(size_t)kr * HD + kc]);
  async_ld_b128(&ldsV[0][vr * 32 + vc], &Vh[(size_t)vr * LSEQ + vc]);

  // Q^T B-fragments (column q = lane): contiguous 16 d values per lane
  v16h qb[2];
  #pragma unroll
  for (int dc = 0; dc < 2; ++dc)
    qb[dc] = frag_b16(Qh + (size_t)(qbase + (lane & 15)) * HD + dc * 32 + hi16);

  v8f ot[4] = {};                 // O^T accumulators: rows d, cols q
  float m = -1e30f, lsum = 0.0f;

  for (int kv = 0; kv < LSEQ; kv += 32) {
    const int buf = (kv >> 5) & 1;
    wait_async0();        // own async stores to LDS landed
    __syncthreads();      // everyone's landed; previous buffer's readers are done
    if (kv + 32 < LSEQ) { // stream next chunk into the other buffer
      async_ld_b128(&ldsK[buf ^ 1][kr * 64 + kc], &Kh[(size_t)(kv + 32 + kr) * HD + kc]);
      async_ld_b128(&ldsV[buf ^ 1][vr * 32 + vc], &Vh[(size_t)vr * LSEQ + kv + 32 + vc]);
    }

    // S^T = K(16kv x 32d) * Q^T(32d x 16q), two kv sub-tiles, two d chunks each
    v8f st0 = {}, st1 = {};
    #pragma unroll
    for (int dc = 0; dc < 2; ++dc) {
      v16h k0 = frag_a16(&ldsK[buf][(0 + (lane & 15)) * 64 + dc * 32 + hi16]);
      st0 = wmma_f16(k0, qb[dc], st0);
      v16h k1 = frag_a16(&ldsK[buf][(16 + (lane & 15)) * 64 + dc * 32 + hi16]);
      st1 = wmma_f16(k1, qb[dc], st1);
    }

    // online softmax: kv lives in register dim + lane half; one shfl_xor(16) completes a row
    F32x8 s0, s1; s0.v = st0; s1.v = st1;
    float mx = s0.f[0];
    #pragma unroll
    for (int r = 1; r < 8; ++r) mx = fmaxf(mx, s0.f[r]);
    #pragma unroll
    for (int r = 0; r < 8; ++r) mx = fmaxf(mx, s1.f[r]);
    mx = fmaxf(mx, __shfl_xor(mx, 16, 32));
    const float mnew  = fmaxf(m, mx);
    const float scale = __expf(m - mnew);
    float rs = 0.0f;
    F16x8 p0, p1;
    #pragma unroll
    for (int r = 0; r < 8; ++r) {
      float e0 = __expf(s0.f[r] - mnew);
      float e1 = __expf(s1.f[r] - mnew);
      rs += e0 + e1;
      p0.h[r] = (_Float16)e0;
      p1.h[r] = (_Float16)e1;
    }
    rs += __shfl_xor(rs, 16, 32);
    lsum = lsum * scale + rs;
    m = mnew;
    #pragma unroll
    for (int dt = 0; dt < 4; ++dt) ot[dt] *= scale;

    // P^T B-fragment: lane-half swap merges the two C tiles, no LDS round-trip
    F16x16 bp;
    const bool lo = lane < 16;
    #pragma unroll
    for (int i = 0; i < 4; ++i) {
      unsigned x0 = __shfl_xor(p0.u[i], 16, 32);
      unsigned x1 = __shfl_xor(p1.u[i], 16, 32);
      bp.u[i]     = lo ? p0.u[i] : x1;
      bp.u[4 + i] = lo ? x0 : p1.u[i];
    }

    // O^T += V^T(16d x 32kv) * P^T(32kv x 16q)
    #pragma unroll
    for (int dt = 0; dt < 4; ++dt) {
      v16h va = frag_a16(&ldsV[buf][(dt * 16 + (lane & 15)) * 32 + hi16]);
      ot[dt] = wmma_f16(va, bp.v, ot[dt]);
    }
  }

  const float inv = 1.0f / lsum;
  const size_t orow = (size_t)nb * LSEQ + qbase + (lane & 15);
  #pragma unroll
  for (int dt = 0; dt < 4; ++dt) {
    F32x8 u; u.v = ot[dt] * inv;
    F16x8 s;
    #pragma unroll
    for (int r = 0; r < 8; ++r) s.h[r] = (_Float16)u.f[r];
    *(uint4*)&O[orow * EMB + head * HD + dt * 16 + hi8] = s.q;
  }
}

// ---------------------------------------------------------------- out = X @ Wo^T + bo
__global__ __launch_bounds__(256) void out_proj(const _Float16* X, const _Float16* Wo16,
                                                const float* bo, float* out) {
  const int lane = threadIdx.x & 31;
  const int wave = threadIdx.x >> 5;
  const int task  = blockIdx.x * 8 + wave;  // 8192 = 512 m-tiles * 16 e-groups
  const int mtile = task >> 4;
  const int ebase = (task & 15) * 64;
  const int hi8  = (lane >> 4) << 3;
  const int hi16 = (lane >> 4) << 4;

  const _Float16* xrow = X + (size_t)(mtile * 16 + (lane & 15)) * EMB;
  v8f acc[4] = {};
  for (int c = 0; c < EMB; c += 32) {
    if (c + 32 < EMB) __builtin_prefetch(xrow + c + 32, 0, 1);
    v16h bf = frag_b16(xrow + c + hi16);
    #pragma unroll
    for (int et = 0; et < 4; ++et) {
      v16h wa = frag_a16(Wo16 + (size_t)(ebase + et * 16 + (lane & 15)) * EMB + c + hi16);
      acc[et] = wmma_f16(wa, bf, acc[et]);
    }
  }
  const size_t mrow = (size_t)mtile * 16 + (lane & 15);
  #pragma unroll
  for (int et = 0; et < 4; ++et) {
    const int e0 = ebase + et * 16 + hi8;
    F32x8 u; u.v = acc[et];
    float4 b0 = *(const float4*)&bo[e0];
    float4 b1 = *(const float4*)&bo[e0 + 4];
    float4 r0 = make_float4(u.f[0] + b0.x, u.f[1] + b0.y, u.f[2] + b0.z, u.f[3] + b0.w);
    float4 r1 = make_float4(u.f[4] + b1.x, u.f[5] + b1.y, u.f[6] + b1.z, u.f[7] + b1.w);
    *(float4*)&out[mrow * EMB + e0]     = r0;
    *(float4*)&out[mrow * EMB + e0 + 4] = r1;
  }
}

extern "C" void kernel_launch(void* const* d_in, const int* in_sizes, int n_in,
                              void* d_out, int out_size, void* d_ws, size_t ws_size,
                              hipStream_t stream) {
  (void)in_sizes; (void)n_in; (void)out_size; (void)ws_size;
  const float* values  = (const float*)d_in[0];
  const float* keys    = (const float*)d_in[1];
  const float* queries = (const float*)d_in[2];
  const float* Wv = (const float*)d_in[3];
  const float* Wk = (const float*)d_in[4];
  const float* Wq = (const float*)d_in[5];
  const float* Wo = (const float*)d_in[6];
  const float* bo = (const float*)d_in[7];
  float* out = (float*)d_out;

  char* ws = (char*)d_ws;
  size_t off = 0;
  _Float16* Wo16 = (_Float16*)(ws + off); off += (size_t)EMB * EMB * 2;
  _Float16* Wv16 = (_Float16*)(ws + off); off += (size_t)HD * HD * 2;
  _Float16* Wk16 = (_Float16*)(ws + off); off += (size_t)HD * HD * 2;
  _Float16* Wq16 = (_Float16*)(ws + off); off += (size_t)HD * HD * 2;
  const size_t tensor_bytes = (size_t)NB * NH * LSEQ * HD * 2;   // 16 MB each
  _Float16* Qp  = (_Float16*)(ws + off); off += tensor_bytes;
  _Float16* Kp  = (_Float16*)(ws + off); off += tensor_bytes;
  _Float16* Vt  = (_Float16*)(ws + off); off += tensor_bytes;
  _Float16* O16 = (_Float16*)(ws + off); off += tensor_bytes;

  cvt_weights<<<(EMB * EMB) / 256, 256, 0, stream>>>(Wv, Wk, Wq, Wo, Wv16, Wk16, Wq16, Wo16);
  proj_qkv<<<1024, 256, 0, stream>>>(queries, keys, values, Wq16, Wk16, Wv16, Qp, Kp, Vt);
  flash_attn<<<dim3(LSEQ / 128, NH, NB), 256, 0, stream>>>(Qp, Kp, Vt, O16);
  out_proj<<<1024, 256, 0, stream>>>(O16, Wo16, bo, out);
}
